// CachedAutoregressiveMixin_51556787421549
// MI455X (gfx1250) — compile-verified, matrix operand order checked by reading, not run on hardware
//
#include <hip/hip_runtime.h>

typedef __attribute__((ext_vector_type(16))) _Float16 v16h;
typedef __attribute__((ext_vector_type(8)))  _Float16 v8h;
typedef __attribute__((ext_vector_type(8)))  float    v8f;
typedef __attribute__((ext_vector_type(4)))  float    f4;

constexpr int CB    = 4;     // batch
constexpr int CNH   = 32;    // heads
constexpr int CSQ   = 512;   // query length
constexpr int CHD   = 128;   // head dim
constexpr int CMLEN = 3584;  // cached length
constexpr int CKL   = CMLEN + CSQ;   // 4096 total keys
constexpr int QT    = 128;   // q rows per workgroup (8 waves x 16)
constexpr int KT    = 32;    // keys per step
constexpr int NSTEPS = CKL / KT;     // 128
constexpr int KSTR  = 136;   // padded f16 stride for K tile rows
constexpr int VSTR  = 40;    // padded f16 stride for V^T rows
constexpr int PSTR  = 40;    // padded f16 stride for P staging

__device__ inline v16h ldv16(const _Float16* p0, const _Float16* p1) {
  union { v16h v; v8h h[2]; } u;
  u.h[0] = *(const v8h*)p0;
  u.h[1] = *(const v8h*)p1;
  return u.v;
}

// DPP16 xor-shuffle within a row of 16 lanes (pure VALU; no LDS round-trip).
// 0xB1 = quad_perm(1,0,3,2)  -> lane ^ 1
// 0x4E = quad_perm(2,3,0,1)  -> lane ^ 2
// 0x164 = ROW_XMASK(4)       -> lane ^ 4
// 0x168 = ROW_XMASK(8)       -> lane ^ 8
template <int CTRL>
__device__ __forceinline__ float dpp_xor_f(float x) {
  int r = __builtin_amdgcn_mov_dpp(__builtin_bit_cast(int, x), CTRL, 0xF, 0xF, true);
  return __builtin_bit_cast(float, r);
}
__device__ __forceinline__ float rowmax16(float t) {
  t = fmaxf(t, dpp_xor_f<0xB1>(t));
  t = fmaxf(t, dpp_xor_f<0x4E>(t));
  t = fmaxf(t, dpp_xor_f<0x164>(t));
  t = fmaxf(t, dpp_xor_f<0x168>(t));
  return t;
}
__device__ __forceinline__ float rowsum16(float t) {
  t += dpp_xor_f<0xB1>(t);
  t += dpp_xor_f<0x4E>(t);
  t += dpp_xor_f<0x164>(t);
  t += dpp_xor_f<0x168>(t);
  return t;
}

__global__ __launch_bounds__(256)
void attn_fa_kernel(const float* __restrict__ q, const float* __restrict__ kk,
                    const float* __restrict__ vv, const float* __restrict__ mask,
                    const float* __restrict__ memp, float* __restrict__ outc) {
  __shared__ __align__(16) _Float16 sK [2][KT][KSTR];   // K tile, row-major (key, hd)
  __shared__ __align__(16) _Float16 sVt[2][CHD][VSTR];  // V tile transposed (hd, key)
  __shared__ __align__(16) _Float16 sP [8][16][PSTR];   // per-wave P staging (D->A relayout)

  const int qtile = blockIdx.x, head = blockIdx.y, b = blockIdx.z;
  const int tid = threadIdx.x, wid = tid >> 5, lane = tid & 31;
  const int l16 = lane & 15, hh = lane >> 4;
  const int qbase = qtile * QT;

  // ---- Q fragments: 16 rows x 128, f32 -> f16 with 1/sqrt(HD) folded in ----
  const float scale = 0.08838834764831845f;
  const int qrow = qbase + wid * 16 + l16;
  const float* qrp = q + (((size_t)b * CNH + head) * CSQ + qrow) * CHD;
  v16h qf[4];
#pragma unroll
  for (int kc = 0; kc < 4; ++kc) {
    const int c0 = kc * 32 + hh * 8;
    f4 a0 = *(const f4*)(qrp + c0);
    f4 a1 = *(const f4*)(qrp + c0 + 4);
    f4 b0 = *(const f4*)(qrp + c0 + 16);
    f4 b1 = *(const f4*)(qrp + c0 + 20);
#pragma unroll
    for (int i = 0; i < 4; ++i) {
      qf[kc][i]      = (_Float16)(a0[i] * scale);
      qf[kc][4 + i]  = (_Float16)(a1[i] * scale);
      qf[kc][8 + i]  = (_Float16)(b0[i] * scale);
      qf[kc][12 + i] = (_Float16)(b1[i] * scale);
    }
  }

  // ---- accumulators / softmax state (lr kept per-lane; reduced in epilogue) ----
  const v8f z8 = {0.f, 0.f, 0.f, 0.f, 0.f, 0.f, 0.f, 0.f};
  v8f o[8];
  float mr[8], lr[8];
#pragma unroll
  for (int i = 0; i < 8; ++i) { o[i] = z8; mr[i] = -3.0e38f; lr[i] = 0.f; }

  // ---- cooperative K/V tile loader: wave w handles rows {w, w+8, w+16, w+24},
  //      lane = float4 column (32 float4 per 128-wide row) ----
  auto tload = [&](int kb, f4* kr, f4* vr) {
#pragma unroll
    for (int i = 0; i < 4; ++i) {
      const int row = wid + i * 8;
      const int key = kb + row;
      const float *ks, *vs;
      if (key < CMLEN) {  // uniform branch: tiles never straddle (3584 % 32 == 0)
        const float* base2 = memp + ((size_t)b * CMLEN + key) * (size_t)(2 * CNH * CHD);
        ks = base2 + (size_t)head * CHD + lane * 4;
        vs = base2 + (size_t)(CNH + head) * CHD + lane * 4;
      } else {
        const int srow = key - CMLEN;
        const size_t off = (((size_t)b * CNH + head) * CSQ + srow) * CHD + lane * 4;
        ks = kk + off; vs = vv + off;
      }
      kr[i] = *(const f4*)ks;
      vr[i] = *(const f4*)vs;
    }
  };
  auto tstore = [&](int st, const f4* kr, const f4* vr) {
#pragma unroll
    for (int i = 0; i < 4; ++i) {
      const int row = wid + i * 8;
      const int c = lane * 4;
      union { _Float16 h[4]; unsigned long long u; } pk;
#pragma unroll
      for (int e = 0; e < 4; ++e) pk.h[e] = (_Float16)kr[i][e];
      *(unsigned long long*)&sK[st][row][c] = pk.u;   // one ds_store_b64
#pragma unroll
      for (int e = 0; e < 4; ++e) sVt[st][c + e][row] = (_Float16)vr[i][e];
    }
  };

  const float* maskb = mask + (size_t)b * CSQ * CKL;
  // per-lane mask base: rows qbase+wid*16+8*hh+r, col kb+l16 (+16)
  const float* mbase = maskb + (size_t)(qbase + wid * 16 + 8 * hh) * CKL + l16;

  // one pipelined step (tile s lives in register set ck/cv; prefetch s+1 into nk/nv)
  auto process = [&](int s, int st, f4* ck, f4* cv, f4* nk_, f4* nv_) {
    const int kb = s * KT;

    tstore(st, ck, cv);                         // convert + stage current tile
    if (s + 1 < NSTEPS) tload(kb + KT, nk_, nv_);  // overlap next global loads
    __syncthreads();                            // tile visible to all waves

    // ---- mask prefetch: one clause of 16 loads, completes under the WMMAs ----
    float m0a[8], m1a[8];
#pragma unroll
    for (int r = 0; r < 8; ++r) {
      const float* mp = mbase + (size_t)r * CKL + kb;
      m0a[r] = mp[0];
      m1a[r] = mp[16];
    }

    // ---- S = Q * K^T  (two 16x16 D tiles: keys kb..kb+15 and kb+16..kb+31) ----
    // fragment loads explicitly pipelined one WMMA ahead (2 live sets -> loose dscnt)
    const _Float16* kb0 = &sK[st][l16][hh * 8];
    const _Float16* kb1 = &sK[st][l16 + 16][hh * 8];
    v8f c0 = z8, c1 = z8;
    v16h ka = ldv16(kb0, kb0 + 16);
    v16h kbf = ldv16(kb1, kb1 + 16);
#pragma unroll
    for (int kc = 0; kc < 4; ++kc) {
      v16h na, nb;
      if (kc < 3) {
        na = ldv16(kb0 + (kc + 1) * 32, kb0 + (kc + 1) * 32 + 16);
        nb = ldv16(kb1 + (kc + 1) * 32, kb1 + (kc + 1) * 32 + 16);
      }
      c0 = __builtin_amdgcn_wmma_f32_16x16x32_f16(false, qf[kc], false, ka,  (short)0, c0, false, false);
      c1 = __builtin_amdgcn_wmma_f32_16x16x32_f16(false, qf[kc], false, kbf, (short)0, c1, false, false);
      if (kc < 3) { ka = na; kbf = nb; }
    }

    // ---- mask + online softmax (row M = r + 8*hh, col N = l16 / l16+16) ----
    // s*m - 1e4*(1-m) == m*(s + 1e4) - 1e4  (add + fma)
#pragma unroll
    for (int r = 0; r < 8; ++r) {
      c0[r] = __builtin_fmaf(m0a[r], c0[r] + 10000.0f, -10000.0f);
      c1[r] = __builtin_fmaf(m1a[r], c1[r] + 10000.0f, -10000.0f);

      const float t = rowmax16(fmaxf(c0[r], c1[r]));
      const float mnew = fmaxf(mr[r], t);
      const float corr = exp2f((mr[r] - mnew) * 1.44269504f);
      mr[r] = mnew;

      const float p0 = exp2f((c0[r] - mnew) * 1.44269504f);
      const float p1 = exp2f((c1[r] - mnew) * 1.44269504f);
      // per-lane partial denominator; corr is lane-uniform within the row,
      // so the 16-lane reduction can be deferred to the epilogue (linearity)
      lr[r] = __builtin_fmaf(lr[r], corr, p0 + p1);
#pragma unroll
      for (int cb = 0; cb < 8; ++cb) o[cb][r] *= corr;

      // D-layout -> LDS (A-layout source); same-wave LDS ops are in-order
      sP[wid][r + 8 * hh][l16]      = (_Float16)p0;
      sP[wid][r + 8 * hh][l16 + 16] = (_Float16)p1;
    }

    // ---- O += P * V  (P: 16x32 A-frag; V^T rows give contiguous B-frags) ----
    v16h pf = ldv16(&sP[wid][l16][hh * 8], &sP[wid][l16][hh * 8 + 16]);
    v16h vf = ldv16(&sVt[st][l16][hh * 8], &sVt[st][l16][hh * 8 + 16]);
#pragma unroll
    for (int cb = 0; cb < 8; ++cb) {
      v16h nv2;
      if (cb < 7) {
        nv2 = ldv16(&sVt[st][(cb + 1) * 16 + l16][hh * 8],
                    &sVt[st][(cb + 1) * 16 + l16][hh * 8 + 16]);
      }
      o[cb] = __builtin_amdgcn_wmma_f32_16x16x32_f16(false, pf, false, vf, (short)0, o[cb], false, false);
      if (cb < 7) vf = nv2;
    }

    __syncthreads();  // all waves done reading buf[st] before it is reused
  };

  f4 rk[4], rv[4], nk[4], nv[4];
  tload(0, rk, rv);

  // unroll-by-2: ping-pong register sets, static stage parity, no reg copies
  for (int s = 0; s < NSTEPS; s += 2) {
    process(s,     0, rk, rv, nk, nv);
    process(s + 1, 1, nk, nv, rk, rv);
  }

  // ---- epilogue: reduce denominators across the row, normalize, store ----
#pragma unroll
  for (int r = 0; r < 8; ++r) {
    const float inv = 1.0f / rowsum16(lr[r]);
    const int row = qbase + wid * 16 + r + 8 * hh;
    float* orow = outc + (((size_t)b * CNH + head) * CSQ + row) * CHD;
#pragma unroll
    for (int cb = 0; cb < 8; ++cb) orow[cb * 16 + l16] = o[cb][r] * inv;
  }
}

// cache_kv: (2,b,nh,sq,hd) stack -> (b,sq,2,nh,hd) flat, float4 copies
__global__ void cachekv_kernel(const float* __restrict__ k, const float* __restrict__ v,
                               float* __restrict__ out) {
  const size_t tot4 = (size_t)CB * CSQ * 2 * CNH * CHD / 4;
  size_t idx = (size_t)blockIdx.x * blockDim.x + threadIdx.x;
  if (idx >= tot4) return;
  const size_t e = idx * 4;
  const int d = (int)(e % CHD);
  size_t t = e / CHD;
  const int h = (int)(t % CNH); t /= CNH;
  const int c = (int)(t % 2);   t /= 2;
  const int s = (int)(t % CSQ);
  const int b = (int)(t / CSQ);
  const float* src = (c == 0 ? k : v) + (((size_t)b * CNH + h) * CSQ + s) * CHD + d;
  *(f4*)(out + e) = *(const f4*)src;
}

extern "C" void kernel_launch(void* const* d_in, const int* in_sizes, int n_in,
                              void* d_out, int out_size, void* d_ws, size_t ws_size,
                              hipStream_t stream) {
  const float* q    = (const float*)d_in[0];
  const float* k    = (const float*)d_in[1];
  const float* v    = (const float*)d_in[2];
  const float* mask = (const float*)d_in[3];
  const float* mem  = (const float*)d_in[4];
  float* ctx   = (float*)d_out;
  float* cache = ctx + (size_t)CB * CNH * CSQ * CHD;

  dim3 grid(CSQ / QT, CNH, CB);   // q-tile fastest -> same-head tiles share K/V in L2
  attn_fa_kernel<<<grid, 256, 0, stream>>>(q, k, v, mask, mem, ctx);

  const size_t tot4 = (size_t)CB * CSQ * 2 * CNH * CHD / 4;
  cachekv_kernel<<<(unsigned)((tot4 + 255) / 256), 256, 0, stream>>>(k, v, cache);
}
